// PraxisSMEAR_38646115729880
// MI455X (gfx1250) — matrix operand order repaired and checked by reading, not compiled, not run on hardware
//
#include <hip/hip_runtime.h>
#include <hip/hip_bf16.h>
#include <math.h>

// Problem constants (from reference)
#define Bb 4
#define Ss 2048
#define Dd 1024
#define Ff 4096
#define Ee 8
#define Mm (Bb * Ss)   // 8192 tokens

typedef __attribute__((ext_vector_type(16))) __bf16 v16bf;
typedef __attribute__((ext_vector_type(8)))  __bf16 v8bf;
typedef __attribute__((ext_vector_type(8)))  float  v8f;
typedef __attribute__((ext_vector_type(4)))  float  v4f;
typedef int vi4 __attribute__((__vector_size__(16)));

#define AS1 __attribute__((address_space(1)))
#define AS3 __attribute__((address_space(3)))
// two-step casts: addrspacecast (proven OK) then same-AS pointer reinterpret
#define GPTR128(p) ((AS1 vi4*)(AS1 void*)(p))
#define LPTR128(p) ((AS3 vi4*)(AS3 void*)(p))

#if __has_builtin(__builtin_amdgcn_global_load_async_to_lds_b128) && \
    __has_builtin(__builtin_amdgcn_s_wait_asynccnt)
#define HAVE_ASYNC_LDS 1
#else
#define HAVE_ASYNC_LDS 0
#endif

// ---------------------------------------------------------------------------
// 1) per-batch column mean of x over the sequence dim:  meanx[b][d]
__global__ __launch_bounds__(256) void colmean_kernel(
    const float* __restrict__ x, float* __restrict__ meanx) {
  int chunks = Dd / 256;
  int b = blockIdx.x / chunks;
  int d = (blockIdx.x % chunks) * 256 + threadIdx.x;
  const float* p = x + (size_t)b * Ss * Dd + d;
  float s = 0.f;
#pragma unroll 4
  for (int i = 0; i < Ss; ++i) s += p[(size_t)i * Dd];
  meanx[b * Dd + d] = s * (1.0f / Ss);
}

// ---------------------------------------------------------------------------
// 2) router: logits = meanx @ router_w + b ; softmax over E ; mean over B
// single wave32: lane = b*8 + e  (4*8 == 32, perfect fit)
__global__ __launch_bounds__(32) void router_kernel(
    const float* __restrict__ meanx, const float* __restrict__ rw,
    const float* __restrict__ rb, float* __restrict__ ew) {
  int lane = threadIdx.x;
  int b = lane >> 3, e = lane & 7;
  float acc = rb[e];
  for (int d = 0; d < Dd; ++d)
    acc = fmaf(meanx[b * Dd + d], rw[d * Ee + e], acc);
  __shared__ float sm[32];
  sm[lane] = acc;
  __syncthreads();
  float mx = -1e30f;
#pragma unroll
  for (int j = 0; j < 8; ++j) mx = fmaxf(mx, sm[b * 8 + j]);
  float ex = __expf(acc - mx);
  __syncthreads();
  sm[lane] = ex;
  __syncthreads();
  float den = 0.f;
#pragma unroll
  for (int j = 0; j < 8; ++j) den += sm[b * 8 + j];
  float prob = ex / den;
  __syncthreads();
  sm[lane] = prob;
  __syncthreads();
  if (lane < 8) {
    float s = 0.f;
#pragma unroll
    for (int bb = 0; bb < Bb; ++bb) s += sm[bb * 8 + lane];
    ew[lane] = s * (1.0f / Bb);
  }
}

// ---------------------------------------------------------------------------
// 3) cast f32 -> bf16 (8 elems / thread); x is streamed once -> NT loads
__global__ __launch_bounds__(256) void cast_bf16_kernel(
    const float* __restrict__ in, __bf16* __restrict__ out, long n) {
  long i = ((long)blockIdx.x * blockDim.x + threadIdx.x) * 8;
  if (i >= n) return;
  v4f f0 = __builtin_nontemporal_load((const v4f*)(in + i));
  v4f f1 = __builtin_nontemporal_load((const v4f*)(in + i + 4));
  v8bf o;
  o[0] = (__bf16)f0.x; o[1] = (__bf16)f0.y; o[2] = (__bf16)f0.z; o[3] = (__bf16)f0.w;
  o[4] = (__bf16)f1.x; o[5] = (__bf16)f1.y; o[6] = (__bf16)f1.z; o[7] = (__bf16)f1.w;
  *(v8bf*)(out + i) = o;
}

// ---------------------------------------------------------------------------
// 4) merged expert weight:  out[i] = bf16( sum_e ew[e] * w[e*n + i] ), 4/thread
//    268 MB streamed exactly once -> non-temporal loads keep L2 for h.
__global__ __launch_bounds__(256) void merge_w_kernel(
    const float* __restrict__ w, const float* __restrict__ ew,
    __bf16* __restrict__ out, long n) {
  long i = ((long)blockIdx.x * blockDim.x + threadIdx.x) * 4;
  if (i >= n) return;
  float s0 = 0.f, s1 = 0.f, s2 = 0.f, s3 = 0.f;
#pragma unroll
  for (int e = 0; e < Ee; ++e) {
    float we = ew[e];
    v4f v = __builtin_nontemporal_load((const v4f*)(w + (long)e * n + i));
    s0 = fmaf(we, v.x, s0); s1 = fmaf(we, v.y, s1);
    s2 = fmaf(we, v.z, s2); s3 = fmaf(we, v.w, s3);
  }
  out[i + 0] = (__bf16)s0; out[i + 1] = (__bf16)s1;
  out[i + 2] = (__bf16)s2; out[i + 3] = (__bf16)s3;
}

// 5) merged bias (f32): bdst[i] = sum_e ew[e]*bsrc[e*n+i]
__global__ __launch_bounds__(256) void merge_b_kernel(
    const float* __restrict__ bsrc, const float* __restrict__ ew,
    float* __restrict__ bdst, int n) {
  int i = blockIdx.x * blockDim.x + threadIdx.x;
  if (i >= n) return;
  float s = 0.f;
#pragma unroll
  for (int e = 0; e < Ee; ++e) s = fmaf(ew[e], bsrc[e * n + i], s);
  bdst[i] = s;
}

// ---------------------------------------------------------------------------
// WMMA bf16 GEMM:  C[M,N] = A[M,K] @ Bw[N,K]^T   (TN: both K-contiguous)
// block: 256 thr = 8 waves (2M x 4N), tile 64M x 256N, K-step 64
// wave tile 32M x 64N = 2x4 v_wmma_f32_16x16x32_bf16 accumulators
// EPI==0: out = bf16(gelu(C + bias))    EPI==1: out = f32(C + bias), NT store
constexpr int BM = 64, BN = 256, BK = 64, LDT = BK + 8;  // LDS stride 72 (16B aligned)

__device__ inline v16bf frag_a(const __bf16* base, int row0, int kk, int lane) {
  // 16-bit A 16x32: lane&15 = M row; lane>>4 selects K-halves {0..7,16..23} / {8..15,24..31}
  int r = lane & 15, kh = (lane >> 4) * 8;
  const __bf16* p = base + (size_t)(row0 + r) * LDT + kk + kh;
  union { v16bf v; v8bf h[2]; } u;
  u.h[0] = *(const v8bf*)p;
  u.h[1] = *(const v8bf*)(p + 16);
  return u.v;
}
__device__ inline v16bf frag_b(const __bf16* base, int nrow0, int kk, int lane) {
  // B 32x16: lane&15 = N col; lane>>4 selects K-halves {0..7,16..23} / {8..15,24..31}
  int n = lane & 15, kh = (lane >> 4) * 8;
  const __bf16* p = base + (size_t)(nrow0 + n) * LDT + kk + kh;
  union { v16bf v; v8bf h[2]; } u;
  u.h[0] = *(const v8bf*)p;
  u.h[1] = *(const v8bf*)(p + 16);
  return u.v;
}
// branch-free tanh-GELU: tanh(u) = copysign((1-e^{-2|u|})/(1+e^{-2|u|}), u)
__device__ inline float gelu_tanh(float x) {
  float u = 0.7978845608028654f * x * fmaf(0.044715f, x * x, 1.0f);
  float a = __expf(-2.0f * fabsf(u));
  float t = (1.0f - a) / (1.0f + a);
  t = copysignf(t, u);
  return 0.5f * x * (1.0f + t);
}

template <int EPI>
__global__ __launch_bounds__(256) void gemm_bf16_kernel(
    const __bf16* __restrict__ A, const __bf16* __restrict__ Bw,
    const float* __restrict__ bias, void* __restrict__ Out,
    int M, int N, int K) {
  __shared__ __bf16 As[BM * LDT];
  __shared__ __bf16 Bs[BN * LDT];

  int t = threadIdx.x;
  int tileN = blockIdx.x * BN;
  int tileM = blockIdx.y * BM;
  int lane = t & 31;
  int wave = t >> 5;
  int mW = (wave & 1) * 32;   // wave M offset in tile
  int nW = (wave >> 1) * 64;  // wave N offset in tile

  v8f acc[2][4] = {};

  int q = t & 7;        // 16B quad within a 128B row
  int rr = t >> 3;      // 0..31

  for (int k0 = 0; k0 < K; k0 += BK) {
    __syncthreads();
#pragma unroll
    for (int j = 0; j < BM / 32; ++j) {
      int r = rr + j * 32;
#if HAVE_ASYNC_LDS
      __builtin_amdgcn_global_load_async_to_lds_b128(
          GPTR128(A + (size_t)(tileM + r) * K + k0 + q * 8),
          LPTR128(&As[r * LDT + q * 8]), 0, 0);
#else
      uint4 v = *(const uint4*)(A + (size_t)(tileM + r) * K + k0 + q * 8);
      *(uint4*)&As[r * LDT + q * 8] = v;
#endif
    }
#pragma unroll
    for (int j = 0; j < BN / 32; ++j) {
      int r = rr + j * 32;
#if HAVE_ASYNC_LDS
      __builtin_amdgcn_global_load_async_to_lds_b128(
          GPTR128(Bw + (size_t)(tileN + r) * K + k0 + q * 8),
          LPTR128(&Bs[r * LDT + q * 8]), 0, 0);
#else
      uint4 v = *(const uint4*)(Bw + (size_t)(tileN + r) * K + k0 + q * 8);
      *(uint4*)&Bs[r * LDT + q * 8] = v;
#endif
    }
#if HAVE_ASYNC_LDS
    __builtin_amdgcn_s_wait_asynccnt(0);
#endif
    __syncthreads();

#pragma unroll
    for (int kk = 0; kk < BK; kk += 32) {
      v16bf af[2], bf[4];
#pragma unroll
      for (int mi = 0; mi < 2; ++mi) af[mi] = frag_a(As, mW + mi * 16, kk, lane);
#pragma unroll
      for (int ni = 0; ni < 4; ++ni) bf[ni] = frag_b(Bs, nW + ni * 16, kk, lane);
#pragma unroll
      for (int mi = 0; mi < 2; ++mi)
#pragma unroll
        for (int ni = 0; ni < 4; ++ni)
          acc[mi][ni] = __builtin_amdgcn_wmma_f32_16x16x32_bf16(
              false, af[mi], false, bf[ni], (short)0, acc[mi][ni], false, false);
    }
  }

  // epilogue: C/D layout -> lane&15 = N, (lane>>4)*8+i = M within 16x16
  int n = lane & 15, mOff = (lane >> 4) * 8;
#pragma unroll
  for (int mi = 0; mi < 2; ++mi) {
#pragma unroll
    for (int ni = 0; ni < 4; ++ni) {
      int col = tileN + nW + ni * 16 + n;
      float bia = bias[col];
#pragma unroll
      for (int i = 0; i < 8; ++i) {
        int row = tileM + mW + mi * 16 + mOff + i;
        float v = acc[mi][ni][i] + bia;
        if constexpr (EPI == 0) {
          ((__bf16*)Out)[(size_t)row * N + col] = (__bf16)gelu_tanh(v);
        } else {
          __builtin_nontemporal_store(v, (float*)Out + (size_t)row * N + col);
        }
      }
    }
  }
}

// ---------------------------------------------------------------------------
extern "C" void kernel_launch(void* const* d_in, const int* in_sizes, int n_in,
                              void* d_out, int out_size, void* d_ws, size_t ws_size,
                              hipStream_t stream) {
  const float* x   = (const float*)d_in[0];
  const float* rw  = (const float*)d_in[1];
  const float* rb  = (const float*)d_in[2];
  const float* w1  = (const float*)d_in[3];
  const float* b1  = (const float*)d_in[4];
  const float* w2  = (const float*)d_in[5];
  const float* b2  = (const float*)d_in[6];
  float* out = (float*)d_out;

  char* w = (char*)d_ws;
  float*  meanx = (float*)w;  w += (size_t)Bb * Dd * sizeof(float);
  float*  ew    = (float*)w;  w += 256;
  float*  mb1   = (float*)w;  w += (size_t)Ff * sizeof(float);
  float*  mb2   = (float*)w;  w += (size_t)Dd * sizeof(float);
  __bf16* xb    = (__bf16*)w; w += (size_t)Mm * Dd * 2;
  __bf16* mw1   = (__bf16*)w; w += (size_t)Ff * Dd * 2;
  __bf16* mw2   = (__bf16*)w; w += (size_t)Dd * Ff * 2;
  __bf16* hb    = (__bf16*)w; w += (size_t)Mm * Ff * 2;

  // 1) router mean over S
  colmean_kernel<<<Bb * (Dd / 256), 256, 0, stream>>>(x, meanx);
  // 2) router logits + softmax + batch mean -> ew[8]
  router_kernel<<<1, 32, 0, stream>>>(meanx, rw, rb, ew);
  // 3) x -> bf16
  {
    long n = (long)Mm * Dd;
    cast_bf16_kernel<<<(int)(n / 8 / 256), 256, 0, stream>>>(x, xb, n);
  }
  // 4) merge expert weights -> bf16
  {
    long n = (long)Ff * Dd;
    merge_w_kernel<<<(int)(n / 4 / 256), 256, 0, stream>>>(w1, ew, mw1, n);
    merge_w_kernel<<<(int)(n / 4 / 256), 256, 0, stream>>>(w2, ew, mw2, n);
  }
  // 5) merge biases
  merge_b_kernel<<<Ff / 256, 256, 0, stream>>>(b1, ew, mb1, Ff);
  merge_b_kernel<<<Dd / 256, 256, 0, stream>>>(b2, ew, mb2, Dd);
  // 6) h = gelu(x @ mw1^T + mb1), bf16
  {
    dim3 g(Ff / BN, Mm / BM);
    gemm_bf16_kernel<0><<<g, 256, 0, stream>>>(xb, mw1, mb1, (void*)hb, Mm, Ff, Dd);
  }
  // 7) out = h @ mw2^T + mb2, f32
  {
    dim3 g(Dd / BN, Mm / BM);
    gemm_bf16_kernel<1><<<g, 256, 0, stream>>>(hb, mw2, mb2, (void*)out, Mm, Dd, Ff);
  }
}
